// Cross_Attention_1726576857078
// MI455X (gfx1250) — matrix operand order repaired and checked by reading, not compiled, not run on hardware
//
#include <hip/hip_runtime.h>
#include <hip/hip_bf16.h>

// ---------------------------------------------------------------------------
// MI455X (gfx1250) cross-attention pipeline.
//
// Roofline: ~240 GFLOP of matmul vs ~300MB HBM traffic (13us @ 23.3TB/s) ->
// compute bound. All GEMMs run as V_WMMA_F32_16X16X32_BF16 (fp32 accumulate).
// All GEMM operands are kept in "NT" layout: A[M,K] row-major, B stored as
// Bt[N,K] row-major (weights already are [out,in]; keys already are [kv,p];
// V is written transposed by the projection epilogue), so both fragment
// loaders read contiguous K.
//
// GEMM shape: 256 threads = 8 waves arranged 4(M) x 2(N); block tile 256x128,
// wave tile 64x64 = 16 WMMA tiles -> 16 WMMAs per 16 b128 loads per K-step.
//
// WMMA wave32 fragment layouts (per CDNA5 ISA 7.12.2):
//   A (16x32 bf16): lane m<16 holds row m, K {0..7,16..23}; lanes 16..31 hold
//                   K {8..15,24..31}  -> two 16B loads per lane.
//   B (32x16 bf16): lane n<16 holds column n, K {0..7,16..23}; lanes 16..31 K {8..15,24..31}
//                   -> two 16B loads per lane from Bt[n, k0+koff].
//   C/D (16x16 f32): VGPR v: lanes 0-15 -> (M=v, N=lane); lanes 16-31 ->
//                   (M=v+8, N=lane-16).
// ---------------------------------------------------------------------------

typedef __bf16 bf16_t;
typedef bf16_t v16bf __attribute__((ext_vector_type(16)));
typedef bf16_t v8bf  __attribute__((ext_vector_type(8)));
typedef bf16_t v4bf  __attribute__((ext_vector_type(4)));
typedef float  v8f   __attribute__((ext_vector_type(8)));

#define BATCH 8
#define SQ    2048
#define SKV   1024
#define EMB_D 1024
#define PROJ_D 1024

// ----------------------------- fp32 -> bf16 --------------------------------
__global__ void cvt_f32_to_bf16(const float* __restrict__ in,
                                bf16_t* __restrict__ out, size_t n4) {
  size_t i = blockIdx.x * (size_t)blockDim.x + threadIdx.x;
  size_t stride = (size_t)gridDim.x * blockDim.x;
  for (; i < n4; i += stride) {
    float4 v = ((const float4*)in)[i];
    v4bf o;
    o[0] = (bf16_t)v.x; o[1] = (bf16_t)v.y;
    o[2] = (bf16_t)v.z; o[3] = (bf16_t)v.w;
    ((v4bf*)out)[i] = o;
  }
}

// --------------------------- WMMA fragment loads ---------------------------
__device__ __forceinline__ v16bf load_a_frag(const bf16_t* __restrict__ p,
                                             int lda, int lane) {
  int m  = lane & 15;
  int ko = (lane & 16) ? 8 : 0;
  const bf16_t* q = p + (size_t)m * lda + ko;
  v8bf lo = *(const v8bf*)(q);        // K = ko .. ko+7
  v8bf hi = *(const v8bf*)(q + 16);   // K = ko+16 .. ko+23
  return __builtin_shufflevector(lo, hi, 0, 1, 2, 3, 4, 5, 6, 7,
                                 8, 9, 10, 11, 12, 13, 14, 15);
}

__device__ __forceinline__ v16bf load_b_frag(const bf16_t* __restrict__ p,
                                             int ldb, int lane) {
  int n  = lane & 15;
  int ko = (lane & 16) ? 8 : 0;
  const bf16_t* q = p + (size_t)n * ldb + ko;
  v8bf lo = *(const v8bf*)(q);        // K = ko .. ko+7
  v8bf hi = *(const v8bf*)(q + 16);   // K = ko+16 .. ko+23
  return __builtin_shufflevector(lo, hi, 0, 1, 2, 3, 4, 5, 6, 7,
                                 8, 9, 10, 11, 12, 13, 14, 15);
}

// ------------------------------ bf16 NT GEMM -------------------------------
// C[b][m][n] = scale * sum_k A[b][m][k] * Bt[b][n][k]  (+ bias[n])
// mode: 0 = f32 row-major, 1 = bf16 row-major, 2 = bf16 transposed (C^T).
__global__ __launch_bounds__(256) void gemm_bf16nt_wmma(
    const bf16_t* __restrict__ A, size_t sA, int lda,
    const bf16_t* __restrict__ Bt, size_t sB, int ldb,
    void* __restrict__ C, size_t sC, int ldc,
    const float* __restrict__ bias, float scale, int mode, int K) {
  const int lane = threadIdx.x & 31;
  const int wave = threadIdx.x >> 5;
  const int bz   = blockIdx.z;
  A  += (size_t)bz * sA;
  Bt += (size_t)bz * sB;
  const size_t cOff = (size_t)bz * sC;

  const int wr0 = blockIdx.y * 256 + (wave >> 1) * 64;  // wave M origin
  const int wn0 = blockIdx.x * 128 + (wave & 1) * 64;   // wave N origin

  v8f acc[4][4];
#pragma unroll
  for (int i = 0; i < 4; ++i)
#pragma unroll
    for (int j = 0; j < 4; ++j)
#pragma unroll
      for (int v = 0; v < 8; ++v) acc[i][j][v] = 0.f;

  const int nk = K >> 5;  // K-steps of 32
  for (int kt = 0; kt < nk; ++kt) {
    const int k0 = kt << 5;
    v16bf bf[4];
#pragma unroll
    for (int j = 0; j < 4; ++j)
      bf[j] = load_b_frag(Bt + (size_t)(wn0 + 16 * j) * ldb + k0, ldb, lane);

    if (kt + 1 < nk) {  // uniform branch: EXEC stays all-ones for WMMA
      __builtin_prefetch(A + (size_t)(wr0 + (lane & 15)) * lda + k0 + 32, 0, 1);
      __builtin_prefetch(Bt + (size_t)(wn0 + (lane & 15)) * ldb + k0 + 32, 0, 1);
    }

#pragma unroll
    for (int i = 0; i < 4; ++i) {
      const v16bf af =
          load_a_frag(A + (size_t)(wr0 + 16 * i) * lda + k0, lda, lane);
#pragma unroll
      for (int j = 0; j < 4; ++j)
        acc[i][j] = __builtin_amdgcn_wmma_f32_16x16x32_bf16(
            false, af, false, bf[j], (short)0, acc[i][j], false, false);
    }
  }

  // epilogue
  const int n  = lane & 15;
  const int mb = (lane >> 4) << 3;  // 0 or 8
#pragma unroll
  for (int i = 0; i < 4; ++i) {
#pragma unroll
    for (int j = 0; j < 4; ++j) {
      const int col = wn0 + 16 * j + n;
      const float bv = bias ? bias[col] : 0.f;
#pragma unroll
      for (int v = 0; v < 8; ++v) {
        const int row = wr0 + 16 * i + mb + v;
        const float val = acc[i][j][v] * scale + bv;
        if (mode == 0)
          ((float*)C)[cOff + (size_t)row * ldc + col] = val;
        else if (mode == 1)
          ((bf16_t*)C)[cOff + (size_t)row * ldc + col] = (bf16_t)val;
        else  // transposed bf16 store (used to materialize V^T)
          ((bf16_t*)C)[cOff + (size_t)col * ldc + row] = (bf16_t)val;
      }
    }
  }
}

// ------------------------------- softmax -----------------------------------
// one wave per row of 1024; in fp32 -> out bf16 probs. scale applied pre-max.
__global__ __launch_bounds__(256) void softmax_rows(
    const float* __restrict__ in, bf16_t* __restrict__ out, float scale) {
  const int lane = threadIdx.x & 31;
  const size_t row = (size_t)blockIdx.x * 8 + (threadIdx.x >> 5);
  const float* p = in + row * 1024;
  float x[32];
#pragma unroll
  for (int i = 0; i < 32; ++i) x[i] = p[i * 32 + lane] * scale;
  float m = -3.402823466e38f;
#pragma unroll
  for (int i = 0; i < 32; ++i) m = fmaxf(m, x[i]);
  for (int off = 16; off; off >>= 1) m = fmaxf(m, __shfl_xor(m, off, 32));
  float s = 0.f;
#pragma unroll
  for (int i = 0; i < 32; ++i) {
    x[i] = __expf(x[i] - m);
    s += x[i];
  }
  for (int off = 16; off; off >>= 1) s += __shfl_xor(s, off, 32);
  const float inv = 1.f / s;
  bf16_t* q = out + row * 1024;
#pragma unroll
  for (int i = 0; i < 32; ++i) q[i * 32 + lane] = (bf16_t)(x[i] * inv);
}

// --------------------------- residual + LayerNorm --------------------------
__global__ __launch_bounds__(256) void add_layernorm(
    const float* __restrict__ obj, const float* __restrict__ attn,
    const float* __restrict__ g, const float* __restrict__ beta,
    float* __restrict__ out) {
  const int lane = threadIdx.x & 31;
  const size_t row = (size_t)blockIdx.x * 8 + (threadIdx.x >> 5);
  const size_t base = row * 1024;
  float x[32];
  float s = 0.f, ss = 0.f;
#pragma unroll
  for (int i = 0; i < 32; ++i) {
    const float v = obj[base + i * 32 + lane] + attn[base + i * 32 + lane];
    x[i] = v;
    s += v;
    ss += v * v;
  }
  for (int off = 16; off; off >>= 1) {
    s += __shfl_xor(s, off, 32);
    ss += __shfl_xor(ss, off, 32);
  }
  const float mu = s * (1.f / 1024.f);
  const float var = ss * (1.f / 1024.f) - mu * mu;
  const float r = rsqrtf(var + 1e-5f);
#pragma unroll
  for (int i = 0; i < 32; ++i) {
    const int c = i * 32 + lane;
    out[base + c] = (x[i] - mu) * r * g[c] + beta[c];
  }
}

// ------------------------------- launcher ----------------------------------
extern "C" void kernel_launch(void* const* d_in, const int* in_sizes, int n_in,
                              void* d_out, int out_size, void* d_ws,
                              size_t ws_size, hipStream_t stream) {
  (void)in_sizes; (void)n_in; (void)out_size; (void)ws_size;

  const float* obj   = (const float*)d_in[0];   // [8,2048,1024]
  const float* sub   = (const float*)d_in[1];   // [8,1024,1024]
  const float* scene = (const float*)d_in[2];   // [8,1024,1024]
  const float* W_q  = (const float*)d_in[3];  const float* b_q  = (const float*)d_in[4];
  const float* W_sk = (const float*)d_in[5];  const float* b_sk = (const float*)d_in[6];
  const float* W_sv = (const float*)d_in[7];  const float* b_sv = (const float*)d_in[8];
  const float* W_ek = (const float*)d_in[9];  const float* b_ek = (const float*)d_in[10];
  const float* W_ev = (const float*)d_in[11]; const float* b_ev = (const float*)d_in[12];
  const float* ln_g = (const float*)d_in[13]; const float* ln_b = (const float*)d_in[14];
  float* out = (float*)d_out;

  const size_t Mq  = (size_t)BATCH * SQ;    // 16384
  const size_t Mkv = (size_t)BATCH * SKV;   // 8192
  const size_t W_ELEM = (size_t)PROJ_D * EMB_D;

  // workspace carve-up (~245 MB total; attn aliases the scores buffer)
  char* ws = (char*)d_ws;
  size_t off = 0;
  auto take = [&](size_t bytes) -> char* {
    char* p = ws + off;
    off += (bytes + 255) & ~(size_t)255;
    return p;
  };
  bf16_t* objh = (bf16_t*)take(Mq * EMB_D * 2);
  bf16_t* subh = (bf16_t*)take(Mkv * EMB_D * 2);
  bf16_t* scnh = (bf16_t*)take(Mkv * EMB_D * 2);
  bf16_t* wqh  = (bf16_t*)take(W_ELEM * 2);
  bf16_t* wskh = (bf16_t*)take(W_ELEM * 2);
  bf16_t* wsvh = (bf16_t*)take(W_ELEM * 2);
  bf16_t* wekh = (bf16_t*)take(W_ELEM * 2);
  bf16_t* wevh = (bf16_t*)take(W_ELEM * 2);
  bf16_t* qh   = (bf16_t*)take(Mq * PROJ_D * 2);
  bf16_t* kh   = (bf16_t*)take(Mkv * PROJ_D * 2);   // reused for sk then ek
  bf16_t* vTh  = (bf16_t*)take(Mkv * PROJ_D * 2);   // [B][P][SKV], reused
  float*  scores = (float*)take(Mq * SKV * 4);      // also reused as attn out
  bf16_t* probs  = (bf16_t*)take(Mq * SKV * 2);
  float*  attn   = scores;  // stream-ordered reuse: scores dead after softmax

  const dim3 blk(256);

  // 1) convert everything to bf16
  cvt_f32_to_bf16<<<dim3(2048), blk, 0, stream>>>(obj, objh, Mq * EMB_D / 4);
  cvt_f32_to_bf16<<<dim3(1024), blk, 0, stream>>>(sub, subh, Mkv * EMB_D / 4);
  cvt_f32_to_bf16<<<dim3(1024), blk, 0, stream>>>(scene, scnh, Mkv * EMB_D / 4);
  cvt_f32_to_bf16<<<dim3(256), blk, 0, stream>>>(W_q, wqh, W_ELEM / 4);
  cvt_f32_to_bf16<<<dim3(256), blk, 0, stream>>>(W_sk, wskh, W_ELEM / 4);
  cvt_f32_to_bf16<<<dim3(256), blk, 0, stream>>>(W_sv, wsvh, W_ELEM / 4);
  cvt_f32_to_bf16<<<dim3(256), blk, 0, stream>>>(W_ek, wekh, W_ELEM / 4);
  cvt_f32_to_bf16<<<dim3(256), blk, 0, stream>>>(W_ev, wevh, W_ELEM / 4);

  // 2) query projection: [16384,1024] @ Wq^T + b_q -> bf16
  gemm_bf16nt_wmma<<<dim3(PROJ_D / 128, Mq / 256, 1), blk, 0, stream>>>(
      objh, 0, EMB_D, wqh, 0, EMB_D, qh, 0, PROJ_D, b_q, 1.f, 1, EMB_D);

  const float scale = 0.03125f;  // PROJ ** -0.5 = 1/32
  const size_t sKVt = (size_t)SKV * PROJ_D;   // per-batch kv stride
  const size_t sQt  = (size_t)SQ * PROJ_D;    // per-batch q stride

  for (int a = 0; a < 2; ++a) {
    const bf16_t* src = a == 0 ? subh : scnh;
    const bf16_t* Wk = a == 0 ? wskh : wekh;
    const bf16_t* Wv = a == 0 ? wsvh : wevh;
    const float* bk = a == 0 ? b_sk : b_ek;
    const float* bv = a == 0 ? b_sv : b_ev;
    float* dst = out + (size_t)a * Mq * PROJ_D;

    // key projection -> kh bf16 row-major [B*SKV, P]
    gemm_bf16nt_wmma<<<dim3(PROJ_D / 128, Mkv / 256, 1), blk, 0, stream>>>(
        src, 0, EMB_D, Wk, 0, EMB_D, kh, 0, PROJ_D, bk, 1.f, 1, EMB_D);

    // value projection, stored transposed per batch -> vTh [B][P][SKV]
    gemm_bf16nt_wmma<<<dim3(PROJ_D / 128, SKV / 256, BATCH), blk, 0, stream>>>(
        src, sKVt, EMB_D, Wv, 0, EMB_D, vTh, sKVt, SKV, bv, 1.f, 2, EMB_D);

    // scores = q @ k^T  (batched; Bt = keys[kv, p] is already NT layout)
    gemm_bf16nt_wmma<<<dim3(SKV / 128, SQ / 256, BATCH), blk, 0, stream>>>(
        qh, sQt, PROJ_D, kh, sKVt, PROJ_D, scores, (size_t)SQ * SKV, SKV,
        nullptr, 1.f, 0, PROJ_D);

    // softmax(scale * scores) -> bf16 probs
    softmax_rows<<<dim3(Mq / 8), blk, 0, stream>>>(scores, probs, scale);

    // attn_out = probs @ v  (Bt = vT[p, kv] is NT layout)
    gemm_bf16nt_wmma<<<dim3(PROJ_D / 128, SQ / 256, BATCH), blk, 0, stream>>>(
        probs, (size_t)SQ * SKV, SKV, vTh, sKVt, SKV, attn,
        (size_t)SQ * PROJ_D, PROJ_D, nullptr, 1.f, 0, SKV);

    // I_a = LayerNorm(obj + attn_out) -> d_out
    add_layernorm<<<dim3(Mq / 8), blk, 0, stream>>>(obj, attn, ln_g, ln_b, dst);
  }
}
